// TransformerXL_962072674620
// MI455X (gfx1250) — matrix operand order repaired and checked by reading, not compiled, not run on hardware
//
#include <hip/hip_runtime.h>
#include <cstddef>
#include <cstdint>

// ---------------- problem constants ----------------
#define TOK   4096     // B*N tokens
#define DIMSZ 1024
#define NSEQ  2048
#define BATCH 2
#define NH    16
#define DHEAD 64
#define DI    2730     // SwiGLU inner
#define N1    5460     // 2*DI
#define N1P   5504     // padded to 64
#define DIP   2752     // DI padded to 32/64

// ---------------- WMMA fragment types ----------------
typedef __attribute__((ext_vector_type(16))) __bf16       v16bf;
typedef __attribute__((ext_vector_type(8)))  float        v8f;
typedef __attribute__((ext_vector_type(4)))  unsigned int u32x4;

union AF {
  v16bf        v;
  u32x4        q[2];
  unsigned int u[8];
};

__device__ __forceinline__ unsigned short f2bf(float f) {
  unsigned int x = __builtin_bit_cast(unsigned int, f);
  x += 0x7fffu + ((x >> 16) & 1u);           // round-to-nearest-even
  return (unsigned short)(x >> 16);
}
__device__ __forceinline__ float bf2f(unsigned short h) {
  unsigned int x = ((unsigned int)h) << 16;
  return __builtin_bit_cast(float, x);
}

// A-fragment (16x32 bf16): lane holds row lane%16; elems 0..7 -> K=8h.., elems 8..15 -> K=16+8h..
__device__ __forceinline__ void load_afrag(AF& f, const unsigned short* row, int k0, int half) {
  f.q[0] = *(const u32x4*)(row + k0 + 8 * half);
  f.q[1] = *(const u32x4*)(row + k0 + 16 + 8 * half);
}
// B-fragment (32x16 bf16, K-contiguous source): elems 0..7 -> K=8h.., elems 8..15 -> K=16+8h..
__device__ __forceinline__ void load_bfrag(AF& f, const unsigned short* row, int k0, int half) {
  f.q[0] = *(const u32x4*)(row + k0 + 8 * half);
  f.q[1] = *(const u32x4*)(row + k0 + 16 + 8 * half);
}
#define WMMA_BF16(a, b, c) \
  __builtin_amdgcn_wmma_f32_16x16x32_bf16(false, (a).v, false, (b).v, (short)0, (c), false, false)

// ---------------- weight convert + transpose (fp32 [K][N] -> bf16 [Np][Kp]) ----------------
__global__ void cvt_tr_kernel(const float* __restrict__ in, unsigned short* __restrict__ out,
                              int K, int N, int Kp, int Np) {
  int idx = blockIdx.x * blockDim.x + threadIdx.x;
  int tot = Np * Kp;
  if (idx >= tot) return;
  int k = idx % Kp;
  int n = idx / Kp;
  float v = (k < K && n < N) ? in[(size_t)k * N + n] : 0.f;
  out[idx] = f2bf(v);
}

// ---------------- RMSNorm: fp32 in, bf16 and/or fp32 out ----------------
__global__ __launch_bounds__(256) void rmsnorm_kernel(const float* __restrict__ x,
                                                      const float* __restrict__ g,
                                                      unsigned short* obf, float* of, int dim) {
  int token = blockIdx.x;
  const float* xr = x + (size_t)token * dim;
  float ss = 0.f;
  for (int i = threadIdx.x; i < dim; i += blockDim.x) { float v = xr[i]; ss += v * v; }
#pragma unroll
  for (int m = 16; m >= 1; m >>= 1) ss += __shfl_xor(ss, m, 32);
  __shared__ float red[8];
  if ((threadIdx.x & 31) == 0) red[threadIdx.x >> 5] = ss;
  __syncthreads();
  float tot = 0.f;
#pragma unroll
  for (int i = 0; i < 8; ++i) tot += red[i];
  float rinv = rsqrtf(tot / (float)dim + 1.1920929e-07f);
  for (int i = threadIdx.x; i < dim; i += blockDim.x) {
    float o = xr[i] * g[i] * rinv;
    if (obf) obf[(size_t)token * dim + i] = f2bf(o);
    if (of)  of[(size_t)token * dim + i] = o;
  }
}

// ---------------- bf16 WMMA GEMM: C[M][N] = A[M][K] @ Bt[N][K]^T (+bias)(+resid) ----------------
// block 256 = 8 waves; wave -> 32(M) x 64(N); block -> 256(M) x 64(N).
// Register double-buffered k-loop (K % 64 == 0) so WMMAs overlap L2 latency.
__global__ __launch_bounds__(256) void gemm_bf16_wmma(const unsigned short* __restrict__ A,
                                                      const unsigned short* __restrict__ Bt,
                                                      const float* __restrict__ bias, int biasN,
                                                      const float* resid, float* C,
                                                      int M, int N, int K) {
  const int lane = threadIdx.x & 31;
  const int w    = threadIdx.x >> 5;
  const int half = lane >> 4;
  const int mrow = lane & 15;
  const int m0 = blockIdx.y * 256 + w * 32;
  const int n0 = blockIdx.x * 64;

  const unsigned short* Ar[2] = { A + (size_t)(m0 + mrow) * K,
                                  A + (size_t)(m0 + 16 + mrow) * K };
  const unsigned short* Br[4] = { Bt + (size_t)(n0 + mrow) * K,
                                  Bt + (size_t)(n0 + 16 + mrow) * K,
                                  Bt + (size_t)(n0 + 32 + mrow) * K,
                                  Bt + (size_t)(n0 + 48 + mrow) * K };

  v8f acc[2][4] = {};
  AF a0[2], a1[2], b0[4], b1[4];
#pragma unroll
  for (int mt = 0; mt < 2; ++mt) load_afrag(a0[mt], Ar[mt], 0, half);
#pragma unroll
  for (int nt = 0; nt < 4; ++nt) load_bfrag(b0[nt], Br[nt], 0, half);

  for (int k0 = 0; k0 < K; k0 += 64) {
    // prefetch k0+32 into buffer 1 (always in range: K % 64 == 0)
#pragma unroll
    for (int mt = 0; mt < 2; ++mt) load_afrag(a1[mt], Ar[mt], k0 + 32, half);
#pragma unroll
    for (int nt = 0; nt < 4; ++nt) load_bfrag(b1[nt], Br[nt], k0 + 32, half);
    // compute on buffer 0 (loads already in flight since previous iteration)
#pragma unroll
    for (int mt = 0; mt < 2; ++mt)
#pragma unroll
      for (int nt = 0; nt < 4; ++nt)
        acc[mt][nt] = WMMA_BF16(a0[mt], b0[nt], acc[mt][nt]);
    if (k0 + 64 < K) {
#pragma unroll
      for (int mt = 0; mt < 2; ++mt) load_afrag(a0[mt], Ar[mt], k0 + 64, half);
#pragma unroll
      for (int nt = 0; nt < 4; ++nt) load_bfrag(b0[nt], Br[nt], k0 + 64, half);
    }
#pragma unroll
    for (int mt = 0; mt < 2; ++mt)
#pragma unroll
      for (int nt = 0; nt < 4; ++nt)
        acc[mt][nt] = WMMA_BF16(a1[mt], b1[nt], acc[mt][nt]);
  }
#pragma unroll
  for (int mt = 0; mt < 2; ++mt)
#pragma unroll
    for (int nt = 0; nt < 4; ++nt)
#pragma unroll
      for (int r = 0; r < 8; ++r) {
        int row = m0 + mt * 16 + r + 8 * half;
        int col = n0 + nt * 16 + mrow;
        float v = acc[mt][nt][r];
        if (bias)  v += (col < biasN) ? bias[col] : 0.f;
        if (resid) v += resid[(size_t)row * N + col];
        C[(size_t)row * N + col] = v;
      }
}

// ---------------- per-head sigmoid projections: gates = sig(t@Wg), mix = sig(t@Wmix) ----------------
__global__ void gates_kernel(const unsigned short* __restrict__ t,
                             const float* __restrict__ Wg, const float* __restrict__ Wm,
                             float* gates, float* mixo, int tokens) {
  int idx = blockIdx.x * blockDim.x + threadIdx.x;
  if (idx >= tokens * NH) return;
  int h = idx & 15, token = idx >> 4;
  const unsigned short* tr = t + (size_t)token * DIMSZ;
  float sg = 0.f, sm = 0.f;
  for (int d = 0; d < DIMSZ; ++d) {
    float tv = bf2f(tr[d]);
    sg += tv * Wg[d * NH + h];
    sm += tv * Wm[d * NH + h];
  }
  gates[idx] = 1.f / (1.f + __expf(-sg));
  mixo[idx]  = 1.f / (1.f + __expf(-sm));
}

// ---------------- RoPE + scale + value-residual lerp; writes bf16 Q,K [B,H,N,DH], Vt [B,H,DH,N] ----------------
__global__ void qkv_post_kernel(const float* __restrict__ qraw, const float* __restrict__ kvraw,
                                const float* __restrict__ mix, float* vres,
                                unsigned short* qr, unsigned short* kr, unsigned short* vt,
                                int nseq, int use_mix) {
  int idx = blockIdx.x * blockDim.x + threadIdx.x;
  int total = BATCH * NH * nseq * 32;
  if (idx >= total) return;
  int pr = idx & 31;
  int n  = (idx >> 5) % nseq;
  int h  = ((idx >> 5) / nseq) % NH;
  int b  = idx / (32 * nseq * NH);
  int token = b * nseq + n;
  int i2 = pr * 2;
  float invf = powf(10000.f, -(float)i2 * (1.f / 64.f));
  float th = (float)n * invf;
  float c = cosf(th), s = sinf(th);
  const float scale = 0.125f;  // 64^-0.5
  size_t qoff = (size_t)token * DIMSZ + h * DHEAD + i2;
  float q0 = qraw[qoff] * scale, q1 = qraw[qoff + 1] * scale;
  size_t ooff = ((size_t)(b * NH + h) * nseq + n) * DHEAD + i2;
  qr[ooff]     = f2bf(q0 * c - q1 * s);
  qr[ooff + 1] = f2bf(q1 * c + q0 * s);
  size_t koff = (size_t)token * (2 * DIMSZ) + h * DHEAD + i2;
  float k0 = kvraw[koff], k1 = kvraw[koff + 1];
  kr[ooff]     = f2bf(k0 * c - k1 * s);
  kr[ooff + 1] = f2bf(k1 * c + k0 * s);
  size_t voff = (size_t)token * (2 * DIMSZ) + DIMSZ + h * DHEAD + i2;
  float v0 = kvraw[voff], v1 = kvraw[voff + 1];
  if (use_mix) {
    float mx = mix[(size_t)token * NH + h];
    v0 = v0 + mx * (vres[ooff] - v0);
    v1 = v1 + mx * (vres[ooff + 1] - v1);
  } else {
    vres[ooff] = v0; vres[ooff + 1] = v1;
  }
  size_t vtoff = ((size_t)(b * NH + h) * DHEAD + i2) * nseq + n;
  vt[vtoff]        = f2bf(v0);
  vt[vtoff + nseq] = f2bf(v1);
}

// ---------------- flash-style windowed attention, 1 wave per 16-query tile ----------------
__global__ __launch_bounds__(32) void attn_kernel(const unsigned short* __restrict__ Qr,
                                                  const unsigned short* __restrict__ Kr,
                                                  const unsigned short* __restrict__ Vt,
                                                  const float* __restrict__ gates,
                                                  unsigned short* __restrict__ Oc, int nseq) {
  const int qs = blockIdx.x * 16;
  const int h = blockIdx.y, b = blockIdx.z;
  const int lane = threadIdx.x;
  const int half = lane >> 4, mrow = lane & 15;
  const unsigned short* Q  = Qr + (size_t)(b * NH + h) * nseq * DHEAD;
  const unsigned short* Kp = Kr + (size_t)(b * NH + h) * nseq * DHEAD;
  const unsigned short* Vp = Vt + (size_t)(b * NH + h) * DHEAD * nseq;
  __shared__ float Pl[16][33];

  AF qa[2];
  {
    const unsigned short* qrow = Q + (size_t)(qs + mrow) * DHEAD;
#pragma unroll
    for (int s2 = 0; s2 < 2; ++s2) load_afrag(qa[s2], qrow, s2 * 32, half);
  }
  v8f acc[4] = {};
  float mi[8], li[8];
#pragma unroll
  for (int r = 0; r < 8; ++r) { mi[r] = -1e29f; li[r] = 0.f; }

  const int j_lo = (qs > 512) ? ((qs - 512) & ~31) : 0;
  const int j_hi = qs + 15;
  for (int j0 = j_lo; j0 <= j_hi; j0 += 32) {
    // issue all 4 K fragments in one clause
    AF kf[2][2];  // [s2][tile]
    const unsigned short* kr0 = Kp + (size_t)(j0 + mrow) * DHEAD;
    const unsigned short* kr1 = Kp + (size_t)(j0 + 16 + mrow) * DHEAD;
#pragma unroll
    for (int s2 = 0; s2 < 2; ++s2) {
      load_bfrag(kf[s2][0], kr0, s2 * 32, half);
      load_bfrag(kf[s2][1], kr1, s2 * 32, half);
    }
    v8f s0 = {}, s1 = {};
    s0 = WMMA_BF16(qa[0], kf[0][0], s0);
    s0 = WMMA_BF16(qa[1], kf[1][0], s0);
    s1 = WMMA_BF16(qa[0], kf[0][1], s1);
    s1 = WMMA_BF16(qa[1], kf[1][1], s1);
    // V fragments are independent of P: issue loads now, they overlap softmax VALU work
    AF vb[4];
#pragma unroll
    for (int nt = 0; nt < 4; ++nt)
      load_bfrag(vb[nt], Vp + (size_t)(nt * 16 + mrow) * nseq + j0, 0, half);
#pragma unroll
    for (int r = 0; r < 8; ++r) {
      int i = qs + r + 8 * half;
      int jA = j0 + mrow, jB = j0 + 16 + mrow;
      float v0 = (jA > i || jA < i - 512) ? -1e30f : s0[r];
      float v1 = (jB > i || jB < i - 512) ? -1e30f : s1[r];
      float mm = fmaxf(v0, v1);
#pragma unroll
      for (int d = 8; d >= 1; d >>= 1) mm = fmaxf(mm, __shfl_xor(mm, d, 32));
      float mn = fmaxf(mi[r], mm);                 // m floor -1e29 keeps masked tiles exact-0
      float alpha = __expf(mi[r] - mn);
      float p0 = __expf(v0 - mn), p1 = __expf(v1 - mn);
      float ls = p0 + p1;
#pragma unroll
      for (int d = 8; d >= 1; d >>= 1) ls += __shfl_xor(ls, d, 32);
      li[r] = li[r] * alpha + ls;
      mi[r] = mn;
#pragma unroll
      for (int nt = 0; nt < 4; ++nt) acc[nt][r] *= alpha;
      Pl[r + 8 * half][mrow]      = p0;            // C-layout -> LDS
      Pl[r + 8 * half][16 + mrow] = p1;
    }
    __syncthreads();                               // single-wave WG: lowers to S_NOP
    AF pa;                                         // rebuild P as A-fragment
#pragma unroll
    for (int p = 0; p < 8; ++p) {
      int kk = (p < 4 ? 2 * p : 2 * p + 8) + 8 * half;
      unsigned int lo = f2bf(Pl[mrow][kk]);
      unsigned int hi = f2bf(Pl[mrow][kk + 1]);
      pa.u[p] = lo | (hi << 16);
    }
    __syncthreads();
#pragma unroll
    for (int nt = 0; nt < 4; ++nt)
      acc[nt] = WMMA_BF16(pa, vb[nt], acc[nt]);
  }
#pragma unroll
  for (int r = 0; r < 8; ++r) {
    int i = qs + r + 8 * half;
    int token = b * nseq + i;
    float sc = gates[(size_t)token * NH + h] / li[r];
#pragma unroll
    for (int nt = 0; nt < 4; ++nt)
      Oc[(size_t)token * DIMSZ + h * DHEAD + nt * 16 + mrow] = f2bf(acc[nt][r] * sc);
  }
}

// ---------------- SwiGLU: s = a * gelu_exact(g), padded bf16 out ----------------
__global__ void swiglu_kernel(const float* __restrict__ u, unsigned short* __restrict__ s, int tokens) {
  int idx = blockIdx.x * blockDim.x + threadIdx.x;
  if (idx >= tokens * DIP) return;
  int c = idx % DIP, token = idx / DIP;
  float val = 0.f;
  if (c < DI) {
    float a = u[(size_t)token * N1P + c];
    float g = u[(size_t)token * N1P + DI + c];
    val = a * (0.5f * g * (1.f + erff(g * 0.70710678118f)));
  }
  s[(size_t)token * DIP + c] = f2bf(val);
}

// ---------------- host driver ----------------
extern "C" void kernel_launch(void* const* d_in, const int* in_sizes, int n_in,
                              void* d_out, int out_size, void* d_ws, size_t ws_size,
                              hipStream_t stream) {
  (void)in_sizes; (void)n_in; (void)out_size; (void)ws_size;
  const float* x_in   = (const float*)d_in[0];
  const float* Wq     = (const float*)d_in[1];
  const float* Wkv    = (const float*)d_in[2];
  const float* Wo     = (const float*)d_in[3];
  const float* Wg     = (const float*)d_in[4];
  const float* Wmix   = (const float*)d_in[5];
  const float* g_attn = (const float*)d_in[6];
  const float* W1     = (const float*)d_in[7];
  const float* b1     = (const float*)d_in[8];
  const float* W2     = (const float*)d_in[9];
  const float* b2     = (const float*)d_in[10];
  const float* g_ff   = (const float*)d_in[11];
  const float* g_fin  = (const float*)d_in[12];

  char* base = (char*)d_ws;
  size_t off = 0;
  auto alloc = [&](size_t bytes) -> void* {
    void* p = base + off;
    off += (bytes + 255) & ~(size_t)255;
    return p;
  };
  float*          xw    = (float*)alloc((size_t)TOK * DIMSZ * 4);
  unsigned short* Tb    = (unsigned short*)alloc((size_t)TOK * DIMSZ * 2);
  float*          qraw  = (float*)alloc((size_t)TOK * DIMSZ * 4);
  float*          kvraw = (float*)alloc((size_t)TOK * 2 * DIMSZ * 4);
  float*          gates = (float*)alloc((size_t)TOK * NH * 4);
  float*          mixb  = (float*)alloc((size_t)TOK * NH * 4);
  unsigned short* qr    = (unsigned short*)alloc((size_t)BATCH * NH * NSEQ * DHEAD * 2);
  unsigned short* kr    = (unsigned short*)alloc((size_t)BATCH * NH * NSEQ * DHEAD * 2 + 16384);
  unsigned short* vt    = (unsigned short*)alloc((size_t)BATCH * NH * DHEAD * NSEQ * 2 + 16384);
  float*          vres  = (float*)alloc((size_t)BATCH * NH * NSEQ * DHEAD * 4);
  unsigned short* oc    = (unsigned short*)alloc((size_t)TOK * DIMSZ * 2);
  float*          ubuf  = (float*)alloc((size_t)TOK * N1P * 4);
  unsigned short* sbuf  = (unsigned short*)alloc((size_t)TOK * DIP * 2);
  unsigned short* WqT   = (unsigned short*)alloc((size_t)4 * DIMSZ * DIMSZ * 2);
  unsigned short* WkvT  = (unsigned short*)alloc((size_t)4 * 2 * DIMSZ * DIMSZ * 2);
  unsigned short* WoT   = (unsigned short*)alloc((size_t)4 * DIMSZ * DIMSZ * 2);
  unsigned short* W1T   = (unsigned short*)alloc((size_t)4 * N1P * DIMSZ * 2);
  unsigned short* W2T   = (unsigned short*)alloc((size_t)4 * DIMSZ * DIP * 2);

  hipMemcpyAsync(xw, x_in, (size_t)TOK * DIMSZ * 4, hipMemcpyDeviceToDevice, stream);

  // weight conversion/transpose: fp32 [K][N] -> bf16 [Np][Kp]; ~100MB bf16 => L2-resident
  for (int l = 0; l < 4; ++l) {
    int n;
    n = DIMSZ * DIMSZ;
    cvt_tr_kernel<<<(n + 255) / 256, 256, 0, stream>>>(Wq + (size_t)l * DIMSZ * DIMSZ,
        WqT + (size_t)l * DIMSZ * DIMSZ, DIMSZ, DIMSZ, DIMSZ, DIMSZ);
    n = 2 * DIMSZ * DIMSZ;
    cvt_tr_kernel<<<(n + 255) / 256, 256, 0, stream>>>(Wkv + (size_t)l * 2 * DIMSZ * DIMSZ,
        WkvT + (size_t)l * 2 * DIMSZ * DIMSZ, DIMSZ, 2 * DIMSZ, DIMSZ, 2 * DIMSZ);
    n = DIMSZ * DIMSZ;
    cvt_tr_kernel<<<(n + 255) / 256, 256, 0, stream>>>(Wo + (size_t)l * DIMSZ * DIMSZ,
        WoT + (size_t)l * DIMSZ * DIMSZ, DIMSZ, DIMSZ, DIMSZ, DIMSZ);
    n = N1P * DIMSZ;
    cvt_tr_kernel<<<(n + 255) / 256, 256, 0, stream>>>(W1 + (size_t)l * DIMSZ * N1,
        W1T + (size_t)l * N1P * DIMSZ, DIMSZ, N1, DIMSZ, N1P);
    n = DIMSZ * DIP;
    cvt_tr_kernel<<<(n + 255) / 256, 256, 0, stream>>>(W2 + (size_t)l * DI * DIMSZ,
        W2T + (size_t)l * DIMSZ * DIP, DI, DIMSZ, DIP, DIMSZ);
  }

  for (int l = 0; l < 4; ++l) {
    rmsnorm_kernel<<<TOK, 256, 0, stream>>>(xw, g_attn + (size_t)l * DIMSZ, Tb, nullptr, DIMSZ);
    gemm_bf16_wmma<<<dim3(DIMSZ / 64, TOK / 256), 256, 0, stream>>>(
        Tb, WqT + (size_t)l * DIMSZ * DIMSZ, nullptr, 0, nullptr, qraw, TOK, DIMSZ, DIMSZ);
    gemm_bf16_wmma<<<dim3(2 * DIMSZ / 64, TOK / 256), 256, 0, stream>>>(
        Tb, WkvT + (size_t)l * 2 * DIMSZ * DIMSZ, nullptr, 0, nullptr, kvraw, TOK, 2 * DIMSZ, DIMSZ);
    gates_kernel<<<(TOK * NH + 255) / 256, 256, 0, stream>>>(
        Tb, Wg + (size_t)l * DIMSZ * NH, Wmix + (size_t)l * DIMSZ * NH, gates, mixb, TOK);
    qkv_post_kernel<<<(BATCH * NH * NSEQ * 32 + 255) / 256, 256, 0, stream>>>(
        qraw, kvraw, mixb, vres, qr, kr, vt, NSEQ, l > 0 ? 1 : 0);
    attn_kernel<<<dim3(NSEQ / 16, NH, BATCH), 32, 0, stream>>>(qr, kr, vt, gates, oc, NSEQ);
    gemm_bf16_wmma<<<dim3(DIMSZ / 64, TOK / 256), 256, 0, stream>>>(
        oc, WoT + (size_t)l * DIMSZ * DIMSZ, nullptr, 0, xw, xw, TOK, DIMSZ, DIMSZ);
    rmsnorm_kernel<<<TOK, 256, 0, stream>>>(xw, g_ff + (size_t)l * DIMSZ, Tb, nullptr, DIMSZ);
    gemm_bf16_wmma<<<dim3(N1P / 64, TOK / 256), 256, 0, stream>>>(
        Tb, W1T + (size_t)l * N1P * DIMSZ, b1 + (size_t)l * N1, N1, nullptr, ubuf, TOK, N1P, DIMSZ);
    swiglu_kernel<<<(TOK * DIP + 255) / 256, 256, 0, stream>>>(ubuf, sbuf, TOK);
    gemm_bf16_wmma<<<dim3(DIMSZ / 64, TOK / 256), 256, 0, stream>>>(
        sbuf, W2T + (size_t)l * DIMSZ * DIP, b2 + (size_t)l * DIMSZ, DIMSZ, xw, xw, TOK, DIMSZ, DIP);
  }
  rmsnorm_kernel<<<TOK, 256, 0, stream>>>(xw, g_fin, nullptr, (float*)d_out, DIMSZ);
}
